// MultiHeadAttention_66803921322083
// MI455X (gfx1250) — matrix operand order repaired and checked by reading, not compiled, not run on hardware
//
#include <hip/hip_runtime.h>
#include <hip/hip_bf16.h>

// ---------------- problem constants ----------------
constexpr int Hn = 16;
constexpr int Mn = 1024;
constexpr int Kn = 64;    // head dim (q/k)
constexpr int Vn = 64;    // head dim (v)
constexpr int Bn = 2;
constexpr int Qn = 2048;
constexpr int Tn = 2048;

typedef __bf16 bf16_t;
typedef __attribute__((ext_vector_type(16))) __bf16 v16bf;
typedef __attribute__((ext_vector_type(8)))  float  v8f;

// ============================================================
// WMMA operand loaders (wave32, 16x16x32 bf16 shapes)
// A (16 rows x 32 k), row-major source with row stride lda:
//   lanes 0-15 : row = lane,    elems 0-7 = K 0-7,   elems 8-15 = K 16-23
//   lanes 16-31: row = lane-16, elems 0-7 = K 8-15,  elems 8-15 = K 24-31
// ============================================================
__device__ __forceinline__ v16bf load_A_bf16(const bf16_t* base, int lda, int lane) {
    int row  = lane & 15;
    int koff = (lane >> 4) * 8;
    const bf16_t* p = base + (long)row * lda + koff;
    v16bf a;
#pragma unroll
    for (int i = 0; i < 8; ++i) { a[i] = p[i]; a[8 + i] = p[16 + i]; }
    return a;
}

// fp32 source (global or LDS after addrspace inference), converted to bf16
__device__ __forceinline__ v16bf load_A_f32(const float* base, int lda, int lane) {
    int row  = lane & 15;
    int koff = (lane >> 4) * 8;
    const float* p = base + (long)row * lda + koff;
    v16bf a;
#pragma unroll
    for (int i = 0; i < 8; ++i) { a[i] = (__bf16)p[i]; a[8 + i] = (__bf16)p[16 + i]; }
    return a;
}

// B (32 k x 16 cols) from "Bt" storage: Bt[col][k] contiguous in k, stride ldb
//   lanes 0-15 : col = lane,    elems 0-7 = K 0-7,   elems 8-15 = K 16-23
//   lanes 16-31: col = lane-16, elems 0-7 = K 8-15,  elems 8-15 = K 24-31
__device__ __forceinline__ v16bf load_Bt_bf16(const bf16_t* bt, int ldb, int lane) {
    int col   = lane & 15;
    int kbase = (lane >> 4) * 8;
    const bf16_t* p = bt + (long)col * ldb + kbase;
    v16bf b;
#pragma unroll
    for (int i = 0; i < 8; ++i) { b[i] = p[i]; b[8 + i] = p[16 + i]; }
    return b;
}

__device__ __forceinline__ v8f wmma_bf16(v16bf a, v16bf b, v8f c) {
    return __builtin_amdgcn_wmma_f32_16x16x32_bf16(false, a, false, b, (short)0, c,
                                                   false, false);
}

// ============================================================
// packing / conversion kernels
// ============================================================
// w: (Hn, Mn, D) fp32 -> wt: (Hn, D, Mn) bf16   (D = 64)
__global__ void mha_pack_wqkv(const float* __restrict__ w, bf16_t* __restrict__ wt, int D) {
    int idx = blockIdx.x * blockDim.x + threadIdx.x;
    int total = Hn * Mn * D;
    if (idx >= total) return;
    int d = idx % D;
    int m = (idx / D) % Mn;
    int h = idx / (D * Mn);
    wt[((long)h * D + d) * Mn + m] = (__bf16)w[idx];
}

// wo: (Hn, Vn, Mn) fp32 -> woT: (Mn, Hn*Vn) bf16
__global__ void mha_pack_wo(const float* __restrict__ wo, bf16_t* __restrict__ woT) {
    int idx = blockIdx.x * blockDim.x + threadIdx.x;
    int total = Hn * Vn * Mn;
    if (idx >= total) return;
    int m  = idx % Mn;
    int hv = idx / Mn;   // h*Vn + v
    woT[(long)m * (Hn * Vn) + hv] = (__bf16)wo[idx];
}

__global__ void mha_cvt_bf16(const float* __restrict__ x, bf16_t* __restrict__ y, int n) {
    int idx = blockIdx.x * blockDim.x + threadIdx.x;
    if (idx < n) y[idx] = (__bf16)x[idx];
}

// ============================================================
// projection: one wave computes a 16x64 block of  X(b) @ W(h)
//   X : (Bn, R, Mn) bf16 row-major ; Wt: (Hn, 64, Mn) bf16 (Bt layout)
//   transposeOut==0: out (Bn,Hn,R,64) bf16 ; ==1: out (Bn,Hn,64,R) bf16
// grid: x = R/16, y = Hn, z = Bn; block = 32 (one wave)
// ============================================================
__global__ void mha_proj(const bf16_t* __restrict__ X, const bf16_t* __restrict__ Wt,
                         bf16_t* __restrict__ out, int R, int transposeOut) {
    int lane = threadIdx.x;
    int rt = blockIdx.x;
    int h  = blockIdx.y;
    int b  = blockIdx.z;

    const bf16_t* Xb = X + ((long)b * R + rt * 16) * Mn;
    const bf16_t* Wh = Wt + (long)h * 64 * Mn;

    v8f acc[4] = {};
    for (int k = 0; k < Mn; k += 32) {
        __builtin_prefetch(Xb + k + 256, 0, 0);
        v16bf a = load_A_bf16(Xb + k, Mn, lane);
#pragma unroll
        for (int dt = 0; dt < 4; ++dt) {
            v16bf bm = load_Bt_bf16(Wh + (long)dt * 16 * Mn + k, Mn, lane);
            acc[dt] = wmma_bf16(a, bm, acc[dt]);
        }
    }

    int col  = lane & 15;
    int rowh = (lane >> 4) * 8;
#pragma unroll
    for (int dt = 0; dt < 4; ++dt) {
        if (!transposeOut) {
            bf16_t* o = out + (((long)(b * Hn + h) * R + rt * 16 + rowh) * 64) + dt * 16 + col;
#pragma unroll
            for (int r = 0; r < 8; ++r) o[(long)r * 64] = (__bf16)acc[dt][r];
        } else {
            bf16_t* o = out + (((long)(b * Hn + h) * 64 + dt * 16 + col) * R) + rt * 16 + rowh;
#pragma unroll
            for (int r = 0; r < 8; ++r) o[r] = (__bf16)acc[dt][r];
        }
    }
}

// ============================================================
// Fused attention: scores -> masked softmax -> PV, one WG per (b,h,16 q-rows).
// 256 threads = 8 waves. att row-block lives in LDS (fp32, padded pitch);
// att is written to HBM exactly once; PV consumes probabilities from LDS.
//   query (Bn,Hn,Qn,64) bf16 ; key (Bn,Hn,Tn,64) bf16 ; valT (Bn,Hn,64,Tn) bf16
//   att (Bn,Hn,Qn,Tn) fp32 out ; pre (Bn,Hn,Qn,64) bf16 out
// grid: x = Hn (fastest -> mask row L2-reuse across heads), y = Qn/16, z = Bn
// ============================================================
constexpr int SPITCH = Tn + 4;   // fp32 pitch; +4 staggers lane halves across banks

__global__ __launch_bounds__(256) void mha_fused_att(
        const bf16_t* __restrict__ query, const bf16_t* __restrict__ key,
        const bf16_t* __restrict__ valT,  const int* __restrict__ mask,
        float* __restrict__ att, bf16_t* __restrict__ pre) {
    __shared__ float S[16 * SPITCH];        // 16 x 2048 logits/probs (padded)
    __shared__ float red[16][17];           // per-row reduction scratch
    __shared__ float pvpart[4][256];        // PV partial tiles (k-half 1)

    int tid  = threadIdx.x;
    int lane = tid & 31;
    int wave = tid >> 5;                    // 0..7
    int h  = blockIdx.x;
    int qt = blockIdx.y;
    int b  = blockIdx.z;
    int bh = b * Hn + h;
    int q0 = qt * 16;

    int col  = lane & 15;
    int rowh = (lane >> 4) * 8;

    // ---------- phase 1: logits into LDS ----------
    const bf16_t* qbase = query + ((long)bh * Qn + q0) * 64;
    v16bf a0 = load_A_bf16(qbase + 0,  64, lane);   // Q tile cached in regs
    v16bf a1 = load_A_bf16(qbase + 32, 64, lane);

    const float kscale = 0.125f;            // 1/sqrt(64)
#pragma unroll 4
    for (int tt = 0; tt < 16; ++tt) {
        int t0 = wave * 256 + tt * 16;
        const bf16_t* kbase = key + ((long)bh * Tn + t0) * 64;
        v8f acc = {};
        acc = wmma_bf16(a0, load_Bt_bf16(kbase + 0,  64, lane), acc);
        acc = wmma_bf16(a1, load_Bt_bf16(kbase + 32, 64, lane), acc);
        float* s = &S[rowh * SPITCH + t0 + col];
#pragma unroll
        for (int r = 0; r < 8; ++r) s[r * SPITCH] = acc[r] * kscale;
    }
    __syncthreads();

    // ---------- phase 2: masked softmax (rows of 2048) ----------
    // thread (row, sub): row = tid>>4 (16 rows), sub = tid&15, 128 elems each
    int row = tid >> 4;
    int sub = tid & 15;
    float* srow = &S[row * SPITCH];
    const int* mrow = mask + ((long)b * Qn + q0 + row) * Tn;

    float lmax = -__builtin_inff();
#pragma unroll 8
    for (int j = 0; j < 128; ++j) {
        int t = sub + (j << 4);
        float x = srow[t];
        if (mrow[t] != 0) { x = -__builtin_inff(); srow[t] = x; }  // inactive
        lmax = fmaxf(lmax, x);
    }
    red[row][sub] = lmax;
    __syncthreads();
    float m = 0.0f;                         // softmax initial=0.0 semantics
#pragma unroll
    for (int i = 0; i < 16; ++i) m = fmaxf(m, red[row][i]);
    __syncthreads();

    float lsum = 0.0f;
#pragma unroll 8
    for (int j = 0; j < 128; ++j) {
        int t = sub + (j << 4);
        float e = __expf(srow[t] - m);      // exp(-inf - m) = 0 for inactive
        srow[t] = e;
        lsum += e;
    }
    red[row][sub] = lsum;
    __syncthreads();
    float sum = 0.0f;
#pragma unroll
    for (int i = 0; i < 16; ++i) sum += red[row][i];
    float inv = 1.0f / sum;

    // normalize in LDS and write att to HBM exactly once
    float* arow = att + ((long)bh * Qn + q0 + row) * Tn;
#pragma unroll 8
    for (int j = 0; j < 128; ++j) {
        int t = sub + (j << 4);
        float p = srow[t] * inv;
        srow[t] = p;
        arow[t] = p;
    }
    __syncthreads();

    // ---------- phase 3: PV from LDS ----------
    // wave -> (d-tile = wave&3, k-half = wave>>2); each half covers T/2 = 1024
    int dt = wave & 3;
    int kh = wave >> 2;
    const bf16_t* vbase = valT + ((long)bh * 64 + dt * 16) * Tn + kh * 1024;
    const float*  abase = &S[kh * 1024];

    v8f acc = {};
#pragma unroll 4
    for (int k = 0; k < 1024; k += 32) {
        v16bf a  = load_A_f32(abase + k, SPITCH, lane);
        v16bf bm = load_Bt_bf16(vbase + k, Tn, lane);
        acc = wmma_bf16(a, bm, acc);
    }

    if (kh == 1) {
#pragma unroll
        for (int r = 0; r < 8; ++r) pvpart[dt][lane * 8 + r] = acc[r];
    }
    __syncthreads();
    if (kh == 0) {
        bf16_t* o = pre + ((long)bh * Qn + q0 + rowh) * 64 + dt * 16 + col;
#pragma unroll
        for (int r = 0; r < 8; ++r) {
            float v = acc[r] + pvpart[dt][lane * 8 + r];
            o[(long)r * 64] = (__bf16)v;
        }
    }
}

// A loader for output projection: A row q, k = h*64 + d over pre(Bn,Hn,Qn,64).
// Each 8-element chunk stays within one head (koff multiples of 8, step 32).
__device__ __forceinline__ v16bf load_A_pre(const bf16_t* pre, int b, int q0, int kk,
                                            int lane) {
    int row  = q0 + (lane & 15);
    int koff = kk + (lane >> 4) * 8;
    v16bf a;
    {
        int h = koff >> 6, d = koff & 63;
        const bf16_t* p = pre + (((long)(b * Hn + h) * Qn + row) * 64) + d;
#pragma unroll
        for (int i = 0; i < 8; ++i) a[i] = p[i];
    }
    {
        int k2 = koff + 16;
        int h = k2 >> 6, d = k2 & 63;
        const bf16_t* p = pre + (((long)(b * Hn + h) * Qn + row) * 64) + d;
#pragma unroll
        for (int i = 0; i < 8; ++i) a[8 + i] = p[i];
    }
    return a;
}

// ============================================================
// out[b,q,m] = pre(b,q, :1024) @ woT ; one wave -> 16x64 output block
// woT: (Mn, 1024) bf16 (Bt layout).  grid: x = Mn/64, y = Qn/16, z = Bn
// ============================================================
__global__ void mha_outproj(const bf16_t* __restrict__ pre, const bf16_t* __restrict__ woT,
                            float* __restrict__ out) {
    int lane = threadIdx.x;
    int mt4 = blockIdx.x;     // 16 groups of 64 m-cols
    int qt  = blockIdx.y;
    int b   = blockIdx.z;

    v8f acc[4] = {};
    for (int kk = 0; kk < Hn * Vn; kk += 32) {
        v16bf a = load_A_pre(pre, b, qt * 16, kk, lane);
#pragma unroll
        for (int mi = 0; mi < 4; ++mi) {
            v16bf bm = load_Bt_bf16(woT + (long)(mt4 * 64 + mi * 16) * (Hn * Vn) + kk,
                                    Hn * Vn, lane);
            acc[mi] = wmma_bf16(a, bm, acc[mi]);
        }
    }

    int col  = lane & 15;
    int rowh = (lane >> 4) * 8;
#pragma unroll
    for (int mi = 0; mi < 4; ++mi) {
        float* o = out + (((long)b * Qn + qt * 16 + rowh) * Mn) + mt4 * 64 + mi * 16 + col;
#pragma unroll
        for (int r = 0; r < 8; ++r) o[(long)r * Mn] = acc[mi][r];
    }
}

// ============================================================
// host launcher
// ============================================================
extern "C" void kernel_launch(void* const* d_in, const int* in_sizes, int n_in,
                              void* d_out, int out_size, void* d_ws, size_t ws_size,
                              hipStream_t stream) {
    const float* qinput  = (const float*)d_in[0];   // (B,Q,M)
    const float* kvinput = (const float*)d_in[1];   // (B,T,M)
    const int*   qtmask  = (const int*)d_in[2];     // (B,Q,T)
    const float* wq      = (const float*)d_in[3];   // (H,M,K)
    const float* wk      = (const float*)d_in[4];   // (H,M,K)
    const float* wv      = (const float*)d_in[5];   // (H,M,V)
    const float* wo      = (const float*)d_in[6];   // (H,V,M)

    float* out = (float*)d_out;                               // (B,Q,M)
    float* att = (float*)d_out + (long)Bn * Qn * Mn;          // (B,H,Q,T)

    // workspace carve-up (bf16 buffers), total ~56 MB
    size_t off = 0;
    auto carve = [&](size_t elems) {
        void* p = (char*)d_ws + off;
        off += elems * sizeof(bf16_t);
        return (bf16_t*)p;
    };
    bf16_t* wqT   = carve((size_t)Hn * 64 * Mn);        // (H,64,M)
    bf16_t* wkT   = carve((size_t)Hn * 64 * Mn);
    bf16_t* wvT   = carve((size_t)Hn * 64 * Mn);
    bf16_t* woT   = carve((size_t)Mn * Hn * Vn);        // (M, H*V)
    bf16_t* qb    = carve((size_t)Bn * Qn * Mn);        // bf16 qinput
    bf16_t* kb    = carve((size_t)Bn * Tn * Mn);        // bf16 kvinput
    bf16_t* query = carve((size_t)Bn * Hn * Qn * 64);   // (B,H,Q,64)
    bf16_t* key   = carve((size_t)Bn * Hn * Tn * 64);   // (B,H,T,64)
    bf16_t* valT  = carve((size_t)Bn * Hn * 64 * Tn);   // (B,H,64,T)
    bf16_t* pre   = carve((size_t)Bn * Hn * Qn * 64);   // (B,H,Q,64)
    (void)ws_size; (void)in_sizes; (void)n_in; (void)out_size;

    // 1) pack weights
    {
        int n = Hn * Mn * 64;
        int blocks = (n + 255) / 256;
        mha_pack_wqkv<<<blocks, 256, 0, stream>>>(wq, wqT, 64);
        mha_pack_wqkv<<<blocks, 256, 0, stream>>>(wk, wkT, 64);
        mha_pack_wqkv<<<blocks, 256, 0, stream>>>(wv, wvT, 64);
        mha_pack_wo<<<blocks, 256, 0, stream>>>(wo, woT);
    }
    // 2) convert activations
    {
        int n = Bn * Qn * Mn;
        mha_cvt_bf16<<<(n + 255) / 256, 256, 0, stream>>>(qinput, qb, n);
        n = Bn * Tn * Mn;
        mha_cvt_bf16<<<(n + 255) / 256, 256, 0, stream>>>(kvinput, kb, n);
    }
    // 3) projections (one wave -> 16x64 block)
    {
        dim3 grid(Qn / 16, Hn, Bn);
        mha_proj<<<grid, 32, 0, stream>>>(qb, wqT, query, Qn, 0);
        dim3 gridT(Tn / 16, Hn, Bn);
        mha_proj<<<gridT, 32, 0, stream>>>(kb, wkT, key, Tn, 0);
        mha_proj<<<gridT, 32, 0, stream>>>(kb, wvT, valT, Tn, 1);
    }
    // 4) fused scores + masked softmax + PV (att written to HBM exactly once)
    {
        dim3 grid(Hn, Qn / 16, Bn);
        mha_fused_att<<<grid, 256, 0, stream>>>(query, key, valT, qtmask, att, pre);
    }
    // 5) output projection (one wave -> 16x64 block)
    {
        dim3 grid(Mn / 64, Qn / 16, Bn);
        mha_outproj<<<grid, 32, 0, stream>>>(pre, woT, out);
    }
}
